// FrequencyBandAttention_9998683865813
// MI455X (gfx1250) — compile-verified, matrix operand order checked by reading, not run on hardware
//
#include <hip/hip_runtime.h>
#include <math.h>

// CDNA5 / gfx1250 fused FrequencyBandAttention forward.
// One block = 16 tokens (144 tok-rows = 9 WMMA row tiles), 256 threads = 8 waves.
// All five GEMMs run on v_wmma_f32_16x16x32_bf16; attention (9x9) is VALU with
// packed-bf16 b128 LDS traffic.

typedef __attribute__((ext_vector_type(16))) __bf16 v16bf;
typedef __attribute__((ext_vector_type(8)))  float  v8f;

#define TOKS_PER_BLK 16
#define NF 9
#define ROWS 144          // 16 tokens * 9 bands
#define DM 128
#define TOK_STRIDE 144    // bf16 elems; 288 B/row (16/32B multiple)
#define QKV_STRIDE 400    // bf16 elems; 800 B/row (16/32B multiple)
#define FLAT_STRIDE 1168  // bf16 elems; 2336 B/row (16/32B multiple)
#define H_STRIDE 272      // bf16 elems; 544 B/row (16/32B multiple)

#define OFF_QKV 41472                 // 144*144*2 bytes for sTok/sCtx region
#define SM_TOTAL (41472 + 115200)     // + 144*400*2 for sQKV/sFlat region

__device__ __forceinline__ v16bf cvt16(const float* __restrict__ p) {
  v16bf r;
#pragma unroll
  for (int i = 0; i < 16; ++i) r[i] = (__bf16)p[i];
  return r;
}

// packed-bf16 (in a u32) -> two floats: bf16 is just the top 16 bits of f32
__device__ __forceinline__ float bflo(unsigned u) {
  union { unsigned u; float f; } c; c.u = u << 16; return c.f;
}
__device__ __forceinline__ float bfhi(unsigned u) {
  union { unsigned u; float f; } c; c.u = u & 0xffff0000u; return c.f;
}

__global__ __launch_bounds__(256) void fba_fused(
    const float* __restrict__ x,
    const float* __restrict__ band_W, const float* __restrict__ band_b,
    const float* __restrict__ freq_pos,
    const float* __restrict__ in_w,  const float* __restrict__ in_b,
    const float* __restrict__ out_w, const float* __restrict__ out_b,
    const float* __restrict__ gate_w, const float* __restrict__ gate_b,
    const float* __restrict__ p1_w,  const float* __restrict__ p1_b,
    const float* __restrict__ p2_w,  const float* __restrict__ p2_b,
    const float* __restrict__ ln_g,  const float* __restrict__ ln_b,
    float* __restrict__ out)
{
  __shared__ __align__(32) unsigned char smem[SM_TOTAL];
  __bf16* sTok  = (__bf16*)smem;                 // phase 0-1
  __bf16* sCtx  = sTok;                          // phase 2-3 (reuse)
  __bf16* sQKV  = (__bf16*)(smem + OFF_QKV);     // phase 1-2
  __bf16* sFlat = sQKV;                          // phase 3-4 (reuse), 16 x 1152
  __bf16* sH    = (__bf16*)smem;                 // phase 4-5 (reuse), 16 x 256
  float*  sGate = (float*)(smem + 8704);         // 16 x 128 f32
  float*  sY    = (float*)(smem + 16896);        // 16 x 128 f32
  float*  sMu   = (float*)(smem + 25088);
  float*  sRs   = (float*)(smem + 25152);

  const int tid  = threadIdx.x;
  const int lane = tid & 31;
  const int wave = tid >> 5;
  const int tok0 = blockIdx.x * TOKS_PER_BLK;
  const int khalf = (lane >> 4) << 4;   // K sub-chunk select for A/B fragments
  const int n     = lane & 15;          // fragment row (A) / column (B)

  // ---------- Phase 0: tok = x*band_W + band_b + freq_pos -> LDS bf16 ----------
  // quad-vectorized: 32 float4 quads per row
  for (int q = tid; q < ROWS * 32; q += 256) {
    int r = q >> 5, c4 = (q & 31) << 2;
    int t = r / NF, f = r - t * NF;
    float xv = x[(tok0 + t) * NF + f];
    const float4 w = *(const float4*)(band_W + f * DM + c4);
    const float4 b = *(const float4*)(band_b + f * DM + c4);
    const float4 p = *(const float4*)(freq_pos + f * DM + c4);
    __bf16 tmp[4];
    tmp[0] = (__bf16)(xv * w.x + b.x + p.x);
    tmp[1] = (__bf16)(xv * w.y + b.y + p.y);
    tmp[2] = (__bf16)(xv * w.z + b.z + p.z);
    tmp[3] = (__bf16)(xv * w.w + b.w + p.w);
    *(uint2*)(sTok + r * TOK_STRIDE + c4) = *(const uint2*)tmp;
  }
  __syncthreads();

  // ---------- Phase 1: qkv(144x384) = tok(144x128) @ in_w^T + in_b ----------
  for (int ct = wave; ct < 24; ct += 8) {       // 24 column tiles of 16
    int col0 = ct * 16;
    int jcol = col0 + n;
    const float* wrow = in_w + jcol * DM + khalf;
    v16bf Bf[4];
#pragma unroll
    for (int kk = 0; kk < 4; ++kk) Bf[kk] = cvt16(wrow + kk * 32);
    float bias = in_b[jcol];
    for (int rt = 0; rt < 9; ++rt) {            // 9 row tiles
      v8f acc = {};
#pragma unroll
      for (int kk = 0; kk < 4; ++kk) {
        v16bf Af = *(const v16bf*)(sTok + (rt * 16 + n) * TOK_STRIDE + kk * 32 + khalf);
        acc = __builtin_amdgcn_wmma_f32_16x16x32_bf16(false, Af, false, Bf[kk],
                                                      (short)0, acc, false, false);
      }
      int mb = rt * 16 + ((lane >> 4) << 3);
#pragma unroll
      for (int i = 0; i < 8; ++i)
        sQKV[(mb + i) * QKV_STRIDE + col0 + n] = (__bf16)(acc[i] + bias);
    }
  }
  __syncthreads();

  // ---------- Phase 2: attention per (token, head, q-row); F=9, H=4, dh=32 ----------
  for (int task = tid; task < TOKS_PER_BLK * 4 * NF; task += 256) {
    int t = task / 36, rem = task - t * 36;
    int h = rem / NF, fq = rem - h * NF;
    const int base = (t * NF) * QKV_STRIDE + h * 32;

    // load q row: 32 bf16 = 4 x b128
    uint4 qu[4];
    {
      const uint4* qp = (const uint4*)(sQKV + base + fq * QKV_STRIDE);
#pragma unroll
      for (int i = 0; i < 4; ++i) qu[i] = qp[i];
    }
    const unsigned* qw = (const unsigned*)qu;

    float s[9], mx = -3.0e38f;
#pragma unroll
    for (int g = 0; g < 9; ++g) {
      uint4 ku[4];
      const uint4* kp = (const uint4*)(sQKV + base + g * QKV_STRIDE + DM);
#pragma unroll
      for (int i = 0; i < 4; ++i) ku[i] = kp[i];
      const unsigned* kw = (const unsigned*)ku;
      float a = 0.f;
#pragma unroll
      for (int i = 0; i < 16; ++i)
        a += bflo(qw[i]) * bflo(kw[i]) + bfhi(qw[i]) * bfhi(kw[i]);
      s[g] = a * 0.17677669529663688f;          // 1/sqrt(32)
      mx = fmaxf(mx, s[g]);
    }
    float sum = 0.f;
#pragma unroll
    for (int g = 0; g < 9; ++g) { s[g] = __expf(s[g] - mx); sum += s[g]; }
    float inv = 1.f / sum;

    float c[32];
#pragma unroll
    for (int i = 0; i < 32; ++i) c[i] = 0.f;
#pragma unroll
    for (int g = 0; g < 9; ++g) {
      uint4 vu[4];
      const uint4* vp = (const uint4*)(sQKV + base + g * QKV_STRIDE + 2 * DM);
#pragma unroll
      for (int i = 0; i < 4; ++i) vu[i] = vp[i];
      const unsigned* vw = (const unsigned*)vu;
      float p = s[g];
#pragma unroll
      for (int i = 0; i < 16; ++i) {
        c[2 * i]     += p * bflo(vw[i]);
        c[2 * i + 1] += p * bfhi(vw[i]);
      }
    }
    __bf16 ob[32];
#pragma unroll
    for (int i = 0; i < 32; ++i) ob[i] = (__bf16)(c[i] * inv);
    uint4* cp = (uint4*)(sCtx + (t * NF + fq) * TOK_STRIDE + h * 32);
    const uint4* op = (const uint4*)ob;
#pragma unroll
    for (int i = 0; i < 4; ++i) cp[i] = op[i];
  }
  __syncthreads();

  // ---------- Phase 3: attn_out = ctx @ out_w^T + out_b -> sFlat(16 x 1152) ----------
  {
    int col0 = wave * 16;                       // 8 column tiles, one per wave
    int jcol = col0 + n;
    const float* wrow = out_w + jcol * DM + khalf;
    v16bf Bf[4];
#pragma unroll
    for (int kk = 0; kk < 4; ++kk) Bf[kk] = cvt16(wrow + kk * 32);
    float bias = out_b[jcol];
    for (int rt = 0; rt < 9; ++rt) {
      v8f acc = {};
#pragma unroll
      for (int kk = 0; kk < 4; ++kk) {
        v16bf Af = *(const v16bf*)(sCtx + (rt * 16 + n) * TOK_STRIDE + kk * 32 + khalf);
        acc = __builtin_amdgcn_wmma_f32_16x16x32_bf16(false, Af, false, Bf[kk],
                                                      (short)0, acc, false, false);
      }
      int mb = rt * 16 + ((lane >> 4) << 3);
#pragma unroll
      for (int i = 0; i < 8; ++i) {
        int row = mb + i;
        int t = row / NF, f = row - t * NF;
        sFlat[t * FLAT_STRIDE + f * DM + jcol] = (__bf16)(acc[i] + bias);
      }
    }
  }
  __syncthreads();

  // ---------- Phase 4: gate(128) & proj1(256) over K=1152, fused ----------
  for (int ct = wave; ct < 24; ct += 8) {       // 0..7 gate, 8..23 proj1
    int col0 = ct * 16;
    bool isGate = (ct < 8);
    const float* wrow = isGate ? (gate_w + (col0 + n) * 1152)
                               : (p1_w + (col0 - 128 + n) * 1152);
    float bias = isGate ? gate_b[col0 + n] : p1_b[col0 - 128 + n];
    v8f acc = {};
    for (int ks = 0; ks < 36; ++ks) {
      v16bf Af = *(const v16bf*)(sFlat + n * FLAT_STRIDE + ks * 32 + khalf);
      v16bf Bf = cvt16(wrow + ks * 32 + khalf);
      acc = __builtin_amdgcn_wmma_f32_16x16x32_bf16(false, Af, false, Bf,
                                                    (short)0, acc, false, false);
    }
    int mb = (lane >> 4) << 3;
    if (isGate) {
#pragma unroll
      for (int i = 0; i < 8; ++i) {
        float v = acc[i] + bias;
        sGate[(mb + i) * DM + col0 + n] = 1.f / (1.f + __expf(-v));
      }
    } else {
#pragma unroll
      for (int i = 0; i < 8; ++i) {
        float v = acc[i] + bias;                // exact GELU
        float g = 0.5f * v * (1.f + erff(v * 0.70710678118654752f));
        sH[(mb + i) * H_STRIDE + (col0 - 128 + n)] = (__bf16)g;
      }
    }
  }
  __syncthreads();

  // ---------- Phase 5: y(16x128) = h(16x256) @ p2_w^T + p2_b ----------
  {
    int col0 = wave * 16;
    const float* wrow = p2_w + (col0 + n) * 256 + khalf;
    float bias = p2_b[col0 + n];
    v8f acc = {};
#pragma unroll
    for (int ks = 0; ks < 8; ++ks) {
      v16bf Af = *(const v16bf*)(sH + n * H_STRIDE + ks * 32 + khalf);
      v16bf Bf = cvt16(wrow + ks * 32);
      acc = __builtin_amdgcn_wmma_f32_16x16x32_bf16(false, Af, false, Bf,
                                                    (short)0, acc, false, false);
    }
    int mb = (lane >> 4) << 3;
#pragma unroll
    for (int i = 0; i < 8; ++i)
      sY[(mb + i) * DM + col0 + n] = acc[i] + bias;
  }
  __syncthreads();

  // ---------- Phase 6: LayerNorm + gate multiply ----------
  if (tid < TOKS_PER_BLK) {
    float mu = 0.f;
    for (int j = 0; j < DM; ++j) mu += sY[tid * DM + j];
    mu *= (1.f / DM);
    float var = 0.f;
    for (int j = 0; j < DM; ++j) { float d = sY[tid * DM + j] - mu; var += d * d; }
    var *= (1.f / DM);
    sMu[tid] = mu;
    sRs[tid] = rsqrtf(var + 1e-5f);
  }
  __syncthreads();
  for (int e = tid; e < TOKS_PER_BLK * DM; e += 256) {
    int t = e >> 7, j = e & 127;
    float v = (sY[e] - sMu[t]) * sRs[t] * ln_g[j] + ln_b[j];
    out[(tok0 + t) * DM + j] = v * sGate[e];
  }
}

extern "C" void kernel_launch(void* const* d_in, const int* in_sizes, int n_in,
                              void* d_out, int out_size, void* d_ws, size_t ws_size,
                              hipStream_t stream) {
  const float* x       = (const float*)d_in[0];
  const float* band_W  = (const float*)d_in[1];
  const float* band_b  = (const float*)d_in[2];
  const float* freqpos = (const float*)d_in[3];
  const float* in_w    = (const float*)d_in[4];
  const float* in_b    = (const float*)d_in[5];
  const float* out_w   = (const float*)d_in[6];
  const float* out_b   = (const float*)d_in[7];
  const float* gate_w  = (const float*)d_in[8];
  const float* gate_b  = (const float*)d_in[9];
  const float* p1_w    = (const float*)d_in[10];
  const float* p1_b    = (const float*)d_in[11];
  const float* p2_w    = (const float*)d_in[12];
  const float* p2_b    = (const float*)d_in[13];
  const float* ln_g    = (const float*)d_in[14];
  const float* ln_b    = (const float*)d_in[15];
  float* out = (float*)d_out;

  int Ntok = in_sizes[0] / NF;              // 32768
  int blocks = Ntok / TOKS_PER_BLK;         // 2048
  fba_fused<<<blocks, 256, 0, stream>>>(x, band_W, band_b, freqpos,
                                        in_w, in_b, out_w, out_b,
                                        gate_w, gate_b, p1_w, p1_b,
                                        p2_w, p2_b, ln_g, ln_b, out);
}
